// CausalWhisperAttention_59072980189926
// MI455X (gfx1250) — matrix operand order repaired and checked by reading, not compiled, not run on hardware
//
#include <hip/hip_runtime.h>
#include <hip/hip_bf16.h>

typedef __attribute__((ext_vector_type(16))) _Float16 v16h;
typedef __attribute__((ext_vector_type(8)))  float    v8f;

#define BATCH 4
#define TLEN  1500
#define DMODEL 1024
#define NHEAD 16
#define HDIM  64
#define MROWS (BATCH * TLEN)   // 6000

// ---------------------------------------------------------------------------
// Generic GEMM: C[m,n] = scale * (sum_k A[m,k] * W[k,n] + bias[n])
// A: M x K (fp32 or f16), W: K x N fp32, C: M x N (f16 or fp32)
// Block: 128 threads = 4 waves. Block tile 128(M) x 64(N). K-step 32.
// Each wave: 32x64 via 8x v_wmma_f32_16x16x32_f16 per K-step.
// W tile is stored TRANSPOSED in LDS so B-fragment reads are contiguous
// (2x ds_load_b128 per fragment instead of 16 scalar ds_load_u16).
// ---------------------------------------------------------------------------
template <typename AT, typename OT>
__global__ __launch_bounds__(128)
void gemm_wmma_kernel(const AT* __restrict__ A, const float* __restrict__ W,
                      const float* __restrict__ bias, OT* __restrict__ Cout,
                      int M, int N, int K, float scale)
{
    __shared__ _Float16 As[128][40];   // 128 rows x 32 k   (pad: 80B row stride)
    __shared__ _Float16 Wst[64][40];   // 64 n x 32 k (transposed W tile)

    const int tid  = threadIdx.x;
    const int wave = tid >> 5;
    const int lane = tid & 31;
    const int ln   = lane & 15;
    const int hi   = lane >> 4;

    const int mBlock = blockIdx.x * 128;
    const int nBlock = blockIdx.y * 64;

    v8f acc[2][4] = {};

    for (int k0 = 0; k0 < K; k0 += 32) {
        // --- cooperative load A tile (128x32): one contiguous row per thread
        {
            int gr = mBlock + tid;
            int cr = (gr < M) ? gr : (M - 1);
            const AT* ap = A + (size_t)cr * K + k0;
            bool ok = (gr < M);
            #pragma unroll
            for (int c = 0; c < 32; ++c) {
                float v = ok ? (float)ap[c] : 0.0f;
                As[tid][c] = (_Float16)v;
            }
            if (k0 + 32 < K)    // prefetch next K-step row
                __builtin_prefetch(ap + 32, 0, 1);
        }
        // --- cooperative load W tile (32x64), store transposed: Wst[n][k]
        {
            int r  = tid >> 2;            // k row within tile (0..31)
            int c0 = (tid & 3) * 16;      // n columns, 16 contiguous
            const float* wp = W + (size_t)(k0 + r) * N + nBlock + c0;
            #pragma unroll
            for (int e = 0; e < 16; ++e)
                Wst[c0 + e][r] = (_Float16)wp[e];
            if (k0 + 32 < K)
                __builtin_prefetch(wp + (size_t)32 * N, 0, 1);
        }
        __syncthreads();

        // A fragments: row m = lane&15, K-set {0..7,16..23} + 8*hi (ISA table)
        v16h af[2];
        #pragma unroll
        for (int u = 0; u < 2; ++u)
            #pragma unroll
            for (int s = 0; s < 16; ++s) {
                int ks = (s < 8 ? s : s + 8) + hi * 8;
                af[u][s] = As[wave * 32 + u * 16 + ln][ks];
            }
        #pragma unroll
        for (int t = 0; t < 4; ++t) {
            // B fragment: col n = lane&15, K-set {0..7,16..23} + 8*hi (ISA table)
            v16h bf;
            #pragma unroll
            for (int s = 0; s < 16; ++s)
                bf[s] = Wst[t * 16 + ln][(s < 8 ? s : s + 8) + hi * 8];
            #pragma unroll
            for (int u = 0; u < 2; ++u)
                acc[u][t] = __builtin_amdgcn_wmma_f32_16x16x32_f16(
                    false, af[u], false, bf, (short)0, acc[u][t], false, false);
        }
        __syncthreads();
    }

    // epilogue: bias + scale; C layout: M = j + 8*hi, N = lane&15
    #pragma unroll
    for (int t = 0; t < 4; ++t) {
        int col = nBlock + t * 16 + ln;
        float bv = bias ? bias[col] : 0.0f;
        #pragma unroll
        for (int u = 0; u < 2; ++u)
            #pragma unroll
            for (int j = 0; j < 8; ++j) {
                int row = mBlock + wave * 32 + u * 16 + j + hi * 8;
                if (row < M) {
                    float v = (acc[u][t][j] + bv) * scale;
                    Cout[(size_t)row * N + col] = (OT)v;
                }
            }
    }
}

// ---------------------------------------------------------------------------
// Attention: per (b, h, 16-query tile), one wave. Online softmax over 32-key
// tiles; mask is a per-row upper bound j <= max(i+50, (i/100)*100+99), so key
// tiles beyond the tile-wide limit are skipped entirely.
// Q/K/V stored f16 as [B*T, H*HD] row-major.
// ---------------------------------------------------------------------------
__global__ __launch_bounds__(128)
void attn_wmma_kernel(const _Float16* __restrict__ Qh,
                      const _Float16* __restrict__ Kh,
                      const _Float16* __restrict__ Vh,
                      _Float16* __restrict__ AO)
{
    __shared__ _Float16 ptile[4][16][34];   // per-wave P transpose staging

    const int tid  = threadIdx.x;
    const int wave = tid >> 5;
    const int lane = tid & 31;
    const int ln   = lane & 15;
    const int hi   = lane >> 4;

    const int b  = blockIdx.z;
    const int h  = blockIdx.y;
    const int qt = blockIdx.x * 4 + wave;
    const int tq0 = qt * 16;
    if (tq0 >= TLEN) return;                 // wave-uniform exit (EXEC stays full)

    const _Float16* Qb = Qh + (size_t)b * TLEN * DMODEL + h * HDIM;
    const _Float16* Kb = Kh + (size_t)b * TLEN * DMODEL + h * HDIM;
    const _Float16* Vb = Vh + (size_t)b * TLEN * DMODEL + h * HDIM;

    // per-row mask limits + softmax stats (rows j + 8*hi, replicated in half-wave)
    int   limit[8];
    float mrow[8], lrow[8];
    #pragma unroll
    for (int j = 0; j < 8; ++j) {
        int i = tq0 + j + hi * 8; if (i > TLEN - 1) i = TLEN - 1;
        int lim = i + 50;
        int ce  = (i / 100) * 100 + 99;
        if (ce > lim) lim = ce;
        if (lim > TLEN - 1) lim = TLEN - 1;
        limit[j] = lim;
        mrow[j] = -1e30f;
        lrow[j] = 0.0f;
    }
    // tile-wide last key (limit is monotonic in i)
    int itop = tq0 + 15; if (itop > TLEN - 1) itop = TLEN - 1;
    int maxkey = itop + 50;
    { int ce = (itop / 100) * 100 + 99; if (ce > maxkey) maxkey = ce; }
    if (maxkey > TLEN - 1) maxkey = TLEN - 1;

    // Q A-fragments for hd slices [0,32) and [32,64)
    v16h qf0, qf1;
    {
        int row = tq0 + ln; if (row >= TLEN) row = TLEN - 1;
        const _Float16* qp = Qb + (size_t)row * DMODEL;
        #pragma unroll
        for (int s = 0; s < 16; ++s) {
            int ks = (s < 8 ? s : s + 8) + hi * 8;
            qf0[s] = qp[ks];
            qf1[s] = qp[32 + ks];
        }
    }

    v8f acc[4] = {};   // 16 x 64 output accumulator, f32

    for (int jb = 0; jb <= maxkey; jb += 32) {
        // ---- S = Q K^T for 32 keys (two 16x16 C-frags, hd=64 -> 2 wmma each)
        v8f sfr[2];
        #pragma unroll
        for (int f = 0; f < 2; ++f) {
            int key = jb + f * 16 + ln;
            int krow = (key >= TLEN) ? TLEN - 1 : key;
            const _Float16* kp = Kb + (size_t)krow * DMODEL;
            v16h kf0, kf1;
            #pragma unroll
            for (int s = 0; s < 16; ++s) {
                int ksb = (s < 8 ? s : s + 8) + hi * 8;
                kf0[s] = kp[ksb];
                kf1[s] = kp[32 + ksb];
            }
            v8f sf = {};
            sf = __builtin_amdgcn_wmma_f32_16x16x32_f16(false, qf0, false, kf0,
                                                        (short)0, sf, false, false);
            sf = __builtin_amdgcn_wmma_f32_16x16x32_f16(false, qf1, false, kf1,
                                                        (short)0, sf, false, false);
            sfr[f] = sf;
        }

        // ---- mask + tile row max
        float tmax[8];
        #pragma unroll
        for (int j = 0; j < 8; ++j) tmax[j] = -1e30f;
        #pragma unroll
        for (int f = 0; f < 2; ++f) {
            int key = jb + f * 16 + ln;
            #pragma unroll
            for (int j = 0; j < 8; ++j) {
                float sv = sfr[f][j];
                sv = (key <= limit[j]) ? sv : -1e30f;   // limit <= T-1 covers OOB keys
                sfr[f][j] = sv;
                tmax[j] = fmaxf(tmax[j], sv);
            }
        }
        #pragma unroll
        for (int off = 1; off < 16; off <<= 1)
            #pragma unroll
            for (int j = 0; j < 8; ++j)
                tmax[j] = fmaxf(tmax[j], __shfl_xor(tmax[j], off, 16));

        // ---- online softmax update
        float corr[8], newm[8], rsum[8];
        #pragma unroll
        for (int j = 0; j < 8; ++j) {
            newm[j] = fmaxf(mrow[j], tmax[j]);
            corr[j] = __expf(mrow[j] - newm[j]);
            mrow[j] = newm[j];
            rsum[j] = 0.0f;
        }
        #pragma unroll
        for (int t = 0; t < 4; ++t)
            #pragma unroll
            for (int j = 0; j < 8; ++j)
                acc[t][j] *= corr[j];

        // ---- P = exp(S - m), write to LDS for layout transpose (C -> A frag)
        #pragma unroll
        for (int f = 0; f < 2; ++f) {
            #pragma unroll
            for (int j = 0; j < 8; ++j) {
                float sv = sfr[f][j];
                float p = (sv > -1e29f) ? __expf(sv - newm[j]) : 0.0f;
                rsum[j] += p;
                ptile[wave][j + hi * 8][f * 16 + ln] = (_Float16)p;
            }
        }
        #pragma unroll
        for (int off = 1; off < 16; off <<= 1)
            #pragma unroll
            for (int j = 0; j < 8; ++j)
                rsum[j] += __shfl_xor(rsum[j], off, 16);
        #pragma unroll
        for (int j = 0; j < 8; ++j)
            lrow[j] = lrow[j] * corr[j] + rsum[j];

        // single wave: ensure LDS stores land before cross-lane gather
        asm volatile("s_wait_dscnt 0" ::: "memory");

        // gather P as A-fragment (row = lane&15, K-set interleaved)
        v16h pf;
        #pragma unroll
        for (int s = 0; s < 16; ++s) {
            int ks = (s < 8 ? s : s + 8) + hi * 8;
            pf[s] = ptile[wave][ln][ks];
        }

        // ---- O += P @ V  (4 n-tiles of 16 hd columns)
        #pragma unroll
        for (int t = 0; t < 4; ++t) {
            v16h vf;
            #pragma unroll
            for (int s = 0; s < 16; ++s) {
                int kk = jb + (s < 8 ? s : s + 8) + hi * 8;
                if (kk >= TLEN) kk = TLEN - 1;       // p == 0 there anyway
                vf[s] = Vb[(size_t)kk * DMODEL + t * 16 + ln];
            }
            acc[t] = __builtin_amdgcn_wmma_f32_16x16x32_f16(
                false, pf, false, vf, (short)0, acc[t], false, false);
        }
    }

    // ---- normalize + store attn output (f16, [B*T, H*HD])
    #pragma unroll
    for (int t = 0; t < 4; ++t) {
        #pragma unroll
        for (int j = 0; j < 8; ++j) {
            int i = tq0 + j + hi * 8;
            if (i < TLEN) {
                float o = acc[t][j] / lrow[j];
                AO[((size_t)b * TLEN + i) * DMODEL + h * HDIM + t * 16 + ln] =
                    (_Float16)o;
            }
        }
    }
}

// ---------------------------------------------------------------------------
extern "C" void kernel_launch(void* const* d_in, const int* in_sizes, int n_in,
                              void* d_out, int out_size, void* d_ws, size_t ws_size,
                              hipStream_t stream)
{
    (void)in_sizes; (void)n_in; (void)out_size; (void)ws_size;

    const float* X  = (const float*)d_in[0];
    const float* Wq = (const float*)d_in[1];
    const float* bq = (const float*)d_in[2];
    const float* Wk = (const float*)d_in[3];
    const float* Wv = (const float*)d_in[4];
    const float* bv = (const float*)d_in[5];
    const float* Wo = (const float*)d_in[6];
    const float* bo = (const float*)d_in[7];
    float* out = (float*)d_out;

    // workspace: Q, K, V, attn-out as f16 [6000 x 1024] each (49.2 MB total)
    const size_t plane = (size_t)MROWS * DMODEL;
    _Float16* Qh = (_Float16*)d_ws;
    _Float16* Kh = Qh + plane;
    _Float16* Vh = Kh + plane;
    _Float16* AO = Vh + plane;

    dim3 blk(128);
    dim3 ggrid((MROWS + 127) / 128, DMODEL / 64);   // 47 x 16

    const float qscale = 0.125f;                    // hd^-0.5 folded into Q
    gemm_wmma_kernel<float, _Float16><<<ggrid, blk, 0, stream>>>(
        X, Wq, bq, Qh, MROWS, DMODEL, DMODEL, qscale);
    gemm_wmma_kernel<float, _Float16><<<ggrid, blk, 0, stream>>>(
        X, Wk, (const float*)nullptr, Kh, MROWS, DMODEL, DMODEL, 1.0f);
    gemm_wmma_kernel<float, _Float16><<<ggrid, blk, 0, stream>>>(
        X, Wv, bv, Vh, MROWS, DMODEL, DMODEL, 1.0f);

    int qtiles = (TLEN + 15) / 16;                  // 94
    dim3 agrid((qtiles + 3) / 4, NHEAD, BATCH);     // 24 x 16 x 4, 4 waves/block
    attn_wmma_kernel<<<agrid, blk, 0, stream>>>(Qh, Kh, Vh, AO);

    gemm_wmma_kernel<_Float16, float><<<ggrid, blk, 0, stream>>>(
        AO, Wo, bo, out, MROWS, DMODEL, DMODEL, 1.0f);
}
